// MultiHeadAttentionWeightSplits_12506944766476
// MI455X (gfx1250) — compile-verified, matrix operand order checked by reading, not run on hardware
//
#include <hip/hip_runtime.h>
#include <hip/hip_bf16.h>

// ---------------------------------------------------------------------------
// MultiHeadAttention forward for MI455X (gfx1250, wave32, WMMA).
//   x:[2,2048,1024] f32, W_q/W_k/W_v:[1024,1024], W_o:[1024,1024], b_o:[1024]
//   out:[2,2048,1024] f32
// Pipeline (f16 compute, f32 accumulate via v_wmma_f32_16x16x32_f16):
//   0) cvt        : x, W_q, W_k, W_v, W_o -> f16 once
//   1) qkv_proj   : A tile staged to LDS via GLOBAL_LOAD_ASYNC_TO_LDS_B128
//                   (ASYNCcnt + s_wait_asynccnt), shared by 8 waves;
//                   Q,K -> [b,h,seq,64] f16 ; V -> transposed [b,h,64,seq] f16
//   2) flash_attn : per (b,h,16-query block), online-softmax streaming over keys
//   3) out_proj   : ctx[token,1024] f16 x W_o^T + b_o -> f32
// Workspace (f16): xh(8MB) W*(8MB) Q(8MB) K(8MB) Vt(8MB) ctx(8MB) = 48MB
// ---------------------------------------------------------------------------

typedef __attribute__((ext_vector_type(16))) _Float16 v16h;
typedef __attribute__((ext_vector_type(8)))  _Float16 v8h;
typedef __attribute__((ext_vector_type(8)))  float    v8f;

#define BSZ      2
#define SEQ      2048
#define D_EMB    1024
#define D_ATTN   1024
#define NHEADS   16
#define HEAD_DIM 64
#define NEG_BIG  (-1.0e30f)

// ---- WMMA fragment helper (layout per CDNA5 ISA 7.12.2) --------------------
// A-fragment (16x32 MxK) gather from a row-major [rows, ld] f16 matrix.
// lane l: row = rbase + (l&15); halves h: K = kbase + (l>>4)*8 + h + (h>=8?8:0)
// => two contiguous 16B chunks at kbase+(l>>4)*8 and +16.
// The same gather against a [N,K]-stored operand produces the B-fragment.
__device__ __forceinline__ v16h frag_f16(const _Float16* __restrict__ p, int ld,
                                         int rbase, int kbase, int lane) {
  const int lo = lane & 15, hi = lane >> 4;
  const _Float16* r = p + (size_t)(rbase + lo) * ld + kbase + hi * 8;
  v8h c0 = *(const v8h*)(r);
  v8h c1 = *(const v8h*)(r + 16);
  v16h out;
#pragma unroll
  for (int i = 0; i < 8; ++i) { out[i] = c0[i]; out[i + 8] = c1[i]; }
  return out;
}

__device__ __forceinline__ v8f wmma16(v16h a, v16h b, v8f c) {
  // D = A(16x32) * B(32x16) + C, f32 accumulate
  return __builtin_amdgcn_wmma_f32_16x16x32_f16(false, a, false, b,
                                                (short)0, c, false, false);
}

// Async copy of one 16B chunk from global memory into LDS (GV mode, ASYNCcnt).
__device__ __forceinline__ void async_g2lds_b128(void* lds_ptr, const void* gptr) {
  const unsigned lds_off = (unsigned)(size_t)lds_ptr;   // addr[31:0] == LDS offset
  const unsigned long long ga = (unsigned long long)(size_t)gptr;
  asm volatile("global_load_async_to_lds_b128 %0, %1, off"
               :: "v"(lds_off), "v"(ga) : "memory");
}
__device__ __forceinline__ void wait_asynccnt0() {
  asm volatile("s_wait_asynccnt 0x0" ::: "memory");
}

// ---------------------------------------------------------------------------
// Kernel 0: bulk f32 -> f16 conversion, 8 elements/thread (b128 in, b128 out).
// ---------------------------------------------------------------------------
__global__ void mha_cvt_f16(const float* __restrict__ src,
                            _Float16* __restrict__ dst, int n8) {
  const int i = blockIdx.x * blockDim.x + threadIdx.x;
  if (i < n8) {
    const float* s = src + (size_t)i * 8;
    v8h o;
#pragma unroll
    for (int j = 0; j < 8; ++j) o[j] = (_Float16)s[j];
    *(v8h*)(dst + (size_t)i * 8) = o;
  }
}

// ---------------------------------------------------------------------------
// Kernel 1: fused QKV projection.  grid = (256, 3), block = 256 (8 waves).
// wave tile nt = by*8 + wave in [0,24): which = nt>>3 (Q/K/V), 8 tiles of 128
// channels each.  The 16x32 A tile (shared by all 8 waves) is staged into LDS
// with async loads once per k-step; each wave runs 8 WMMAs against preloaded
// B-fragments (deep load pipeline).  K-loop trip count is block-uniform, so
// block barriers are safe.
// ---------------------------------------------------------------------------
__global__ void mha_qkv_proj(const _Float16* __restrict__ xh,
                             const _Float16* __restrict__ Wq,
                             const _Float16* __restrict__ Wk,
                             const _Float16* __restrict__ Wv,
                             _Float16* __restrict__ Qh,
                             _Float16* __restrict__ Kh,
                             _Float16* __restrict__ Vt) {
  __shared__ _Float16 Atile[16][32];   // 1 KB staging for the current k-step

  const int tid  = threadIdx.x;
  const int lane = tid & 31;
  const int wave = tid >> 5;
  const int tok0 = blockIdx.x * 16;
  const int nt   = blockIdx.y * 8 + wave;          // 0..23
  const int which = nt >> 3;                       // 0=Q,1=K,2=V
  const int n0    = (nt & 7) * 128;                // channel base within matrix
  const _Float16* __restrict__ W = (which == 0) ? Wq : (which == 1) ? Wk : Wv;

  const int lo = lane & 15, hi = lane >> 4;

  // This thread's staging assignment: 64 chunks of 16B (16 rows x 4 chunks).
  const int srow = tid >> 2, schk = (tid & 3) * 8;

  v8f acc[8] = {};
  for (int k0 = 0; k0 < D_EMB; k0 += 32) {
    // ---- stage A tile (16 tokens x 32 k) into LDS, async ----------------
    if (tid < 64) {
      async_g2lds_b128(&Atile[srow][schk],
                       xh + (size_t)(tok0 + srow) * D_EMB + k0 + schk);
    }
    wait_asynccnt0();
    __syncthreads();

    const v16h a = frag_f16(&Atile[0][0], 32, 0, 0, lane);  // ds_load_b128 x2

    // ---- preload all 8 B-fragments, then 8 WMMAs (loads overlap WMMAs) --
    v16h bfr[8];
#pragma unroll
    for (int t = 0; t < 8; ++t)
      bfr[t] = frag_f16(W, D_EMB, n0 + t * 16, k0, lane);
#pragma unroll
    for (int t = 0; t < 8; ++t)
      acc[t] = wmma16(a, bfr[t], acc[t]);

    __syncthreads();   // protect Atile before next k-step overwrites it
  }

#pragma unroll
  for (int t = 0; t < 8; ++t) {
    const int ch = n0 + t * 16 + lo;
    const int h  = ch >> 6, hd = ch & 63;
#pragma unroll
    for (int j = 0; j < 8; ++j) {
      const int tok = tok0 + j + hi * 8;
      const int bb  = tok >> 11, s = tok & (SEQ - 1);
      const _Float16 v = (_Float16)acc[t][j];
      if (which == 2) {
        // V transposed: [b, h, hd, seq] -> contiguous keys for P*V B-fragments
        Vt[(((size_t)bb * NHEADS + h) * HEAD_DIM + hd) * SEQ + s] = v;
      } else {
        _Float16* __restrict__ dst = (which == 0) ? Qh : Kh;
        dst[(((size_t)bb * NHEADS + h) * SEQ + s) * HEAD_DIM + hd] = v;
      }
    }
  }
}

// ---------------------------------------------------------------------------
// Kernel 2: flash attention.  grid = (SEQ/16, NHEADS, BSZ), block = 32 (1 wave,
// so barriers are S_NOPs and the causally-varying key loop is safe).
// One 16-query tile vs all keys <= q_max, online softmax, O in 4 C-fragments.
// ---------------------------------------------------------------------------
__global__ void mha_flash_attn(const _Float16* __restrict__ Qh,
                               const _Float16* __restrict__ Kh,
                               const _Float16* __restrict__ Vt,
                               _Float16* __restrict__ ctx) {
  __shared__ float    Sf[16][32];
  __shared__ _Float16 Pt[16][32];
  __shared__ float    rowfac[16];
  __shared__ float    rowsum[16];

  const int lane = threadIdx.x & 31;
  const int lo = lane & 15, hi = lane >> 4;
  const int q0 = blockIdx.x * 16;
  const int h  = blockIdx.y;
  const int bb = blockIdx.z;

  const _Float16* __restrict__ Qb = Qh + ((size_t)bb * NHEADS + h) * SEQ * HEAD_DIM;
  const _Float16* __restrict__ Kb = Kh + ((size_t)bb * NHEADS + h) * SEQ * HEAD_DIM;
  const _Float16* __restrict__ Vb = Vt + ((size_t)bb * NHEADS + h) * HEAD_DIM * SEQ;

  // Q A-fragments (16 rows x 64 head-dim) -- loaded once, reused every block.
  const v16h aq0 = frag_f16(Qb, HEAD_DIM, q0, 0, lane);
  const v16h aq1 = frag_f16(Qb, HEAD_DIM, q0, 32, lane);

  v8f o[4] = {};
  float mrow = NEG_BIG, lrow = 0.0f;   // stats for row (lane&15), dup in lane^16

  const float scale = 0.125f;          // 1/sqrt(64)

  for (int k0 = 0; k0 < q0 + 16; k0 += 32) {
    // ---- S = Q * K^T for 32 keys (two 16-key N-tiles) --------------------
#pragma unroll
    for (int n = 0; n < 2; ++n) {
      v16h bk0 = frag_f16(Kb, HEAD_DIM, k0 + n * 16, 0, lane);
      v16h bk1 = frag_f16(Kb, HEAD_DIM, k0 + n * 16, 32, lane);
      v8f sv = {};
      sv = wmma16(aq0, bk0, sv);
      sv = wmma16(aq1, bk1, sv);
      const int col = n * 16 + lo;     // key column in this 16x32 tile
#pragma unroll
      for (int j = 0; j < 8; ++j) {
        const int row = j + hi * 8;
        float val = sv[j] * scale;
        if (k0 + col > q0 + row) val = NEG_BIG;   // causal mask
        Sf[row][col] = val;
      }
    }
    __syncthreads();

    // ---- online softmax over the 16x32 tile ------------------------------
    {
      const int r = lo;                 // lanes (r) and (r+16) cooperate on row r
      float lm = NEG_BIG;
#pragma unroll
      for (int c = 0; c < 16; ++c) lm = fmaxf(lm, Sf[r][hi * 16 + c]);
      lm = fmaxf(lm, __shfl_xor(lm, 16, 32));
      const float mnew = fmaxf(mrow, lm);
      float ps = 0.0f;
#pragma unroll
      for (int c = 0; c < 16; ++c) {
        const float p = __expf(Sf[r][hi * 16 + c] - mnew);
        Pt[r][hi * 16 + c] = (_Float16)p;
        ps += p;
      }
      ps += __shfl_xor(ps, 16, 32);
      const float fac = __expf(mrow - mnew);
      lrow = lrow * fac + ps;
      mrow = mnew;
      if (lane < 16) rowfac[r] = fac;
    }
    __syncthreads();

    // ---- rescale O, then O += P * V --------------------------------------
#pragma unroll
    for (int j = 0; j < 8; ++j) {
      const float f = rowfac[j + hi * 8];
#pragma unroll
      for (int t = 0; t < 4; ++t) o[t][j] *= f;
    }
    const v16h ap = frag_f16(&Pt[0][0], 32, 0, 0, lane);
#pragma unroll
    for (int t = 0; t < 4; ++t) {
      v16h bv = frag_f16(Vb, SEQ, t * 16, k0, lane);  // V^T rows=hd, cols=keys
      o[t] = wmma16(ap, bv, o[t]);
    }
    __syncthreads();
  }

  if (lane < 16) rowsum[lo] = lrow;
  __syncthreads();

  // ---- write ctx[token, 1024] f16 ---------------------------------------
#pragma unroll
  for (int j = 0; j < 8; ++j) {
    const int row = j + hi * 8;
    const float inv = 1.0f / rowsum[row];
    const size_t tok = (size_t)bb * SEQ + q0 + row;
#pragma unroll
    for (int t = 0; t < 4; ++t) {
      const int ch = h * HEAD_DIM + t * 16 + lo;
      ctx[tok * D_ATTN + ch] = (_Float16)(o[t][j] * inv);
    }
  }
}

// ---------------------------------------------------------------------------
// Kernel 3: output projection + bias.  grid = (256, 1), block = 256 (8 waves).
// wave n0 = wave*128: 16 tokens x 128 output channels per wave, preloaded
// B-fragment pipeline as in kernel 1.
// ---------------------------------------------------------------------------
__global__ void mha_out_proj(const _Float16* __restrict__ ctx,
                             const _Float16* __restrict__ Wo,
                             const float* __restrict__ bo,
                             float* __restrict__ out) {
  const int lane = threadIdx.x & 31;
  const int wave = threadIdx.x >> 5;
  const int tok0 = blockIdx.x * 16;
  const int n0   = wave * 128;

  const int lo = lane & 15, hi = lane >> 4;

  v8f acc[8] = {};
  for (int k0 = 0; k0 < D_ATTN; k0 += 32) {
    v16h a = frag_f16(ctx, D_ATTN, tok0, k0, lane);
    if (k0 + 32 < D_ATTN) {
      __builtin_prefetch(ctx + (size_t)(tok0 + lo) * D_ATTN + k0 + 32 + hi * 8, 0, 1);
    }
    v16h bfr[8];
#pragma unroll
    for (int t = 0; t < 8; ++t)
      bfr[t] = frag_f16(Wo, D_ATTN, n0 + t * 16, k0, lane);
#pragma unroll
    for (int t = 0; t < 8; ++t)
      acc[t] = wmma16(a, bfr[t], acc[t]);
  }

#pragma unroll
  for (int t = 0; t < 8; ++t) {
    const int ch = n0 + t * 16 + lo;
    const float bias = bo[ch];
#pragma unroll
    for (int j = 0; j < 8; ++j) {
      const size_t tok = tok0 + j + hi * 8;
      out[tok * D_ATTN + ch] = acc[t][j] + bias;
    }
  }
}

// ---------------------------------------------------------------------------
extern "C" void kernel_launch(void* const* d_in, const int* in_sizes, int n_in,
                              void* d_out, int out_size, void* d_ws, size_t ws_size,
                              hipStream_t stream) {
  const float* x   = (const float*)d_in[0];
  const float* W_q = (const float*)d_in[1];
  const float* W_k = (const float*)d_in[2];
  const float* W_v = (const float*)d_in[3];
  const float* W_o = (const float*)d_in[4];
  const float* b_o = (const float*)d_in[5];
  float* out = (float*)d_out;

  // Workspace layout (f16 halfs):
  //   xh (4M) | Wq (1M) | Wk (1M) | Wv (1M) | Wo (1M) | Q (4M) | K (4M)
  //   | Vt (4M) | ctx (4M)   => 24M halfs = 48 MB
  _Float16* wsh = (_Float16*)d_ws;
  const size_t NX = (size_t)BSZ * SEQ * D_EMB;          // 4,194,304
  const size_t NW = (size_t)D_ATTN * D_EMB;             // 1,048,576
  _Float16* xh  = wsh;
  _Float16* Wqh = xh + NX;
  _Float16* Wkh = Wqh + NW;
  _Float16* Wvh = Wkh + NW;
  _Float16* Wvx = Wvh;  (void)Wvx;
  _Float16* Woh = Wvh + NW;
  _Float16* Qh  = Woh + NW;
  _Float16* Kh  = Qh + NX;
  _Float16* Vt  = Kh + NX;
  _Float16* ctx = Vt + NX;

  // 0) One-time f32 -> f16 conversion (each element converted exactly once).
  mha_cvt_f16<<<(int)(NX / 8 / 256), 256, 0, stream>>>(x, xh, (int)(NX / 8));
  mha_cvt_f16<<<(int)(NW / 8 / 256), 256, 0, stream>>>(W_q, Wqh, (int)(NW / 8));
  mha_cvt_f16<<<(int)(NW / 8 / 256), 256, 0, stream>>>(W_k, Wkh, (int)(NW / 8));
  mha_cvt_f16<<<(int)(NW / 8 / 256), 256, 0, stream>>>(W_v, Wvh, (int)(NW / 8));
  mha_cvt_f16<<<(int)(NW / 8 / 256), 256, 0, stream>>>(W_o, Woh, (int)(NW / 8));

  const int TOKTILES = (BSZ * SEQ) / 16;   // 256

  // 1) QKV projection: 24 channel tiles of 128 over {Q,K,V}, 8 waves/block.
  mha_qkv_proj<<<dim3(TOKTILES, 3), 256, 0, stream>>>(xh, Wqh, Wkh, Wvh, Qh, Kh, Vt);

  // 2) Flash attention: one wave per (16-query block, head, batch).
  mha_flash_attn<<<dim3(SEQ / 16, NHEADS, BSZ), 32, 0, stream>>>(Qh, Kh, Vt, ctx);

  // 3) Output projection + bias.
  mha_out_proj<<<TOKTILES, 256, 0, stream>>>(ctx, Woh, b_o, out);
}
